// VisualSingleModule_5050881540240
// MI455X (gfx1250) — matrix operand order repaired and checked by reading, not compiled, not run on hardware
//
#include <hip/hip_runtime.h>
#include <hip/hip_bf16.h>

typedef _Float16 half_t;
typedef __attribute__((ext_vector_type(16))) _Float16 v16h;
typedef __attribute__((ext_vector_type(8)))  float    v8f;

#define HH     64
#define WWID   64
#define PADB   2
#define PH     68
#define PW     68
#define CINC   64
#define BBATCH 16
#define KSLOT  6
#define DDIM   64
#define NIMG   96          // B*K
#define KTOT   1600        // 25 taps * 64 cin
#define NFRAME 6

// ---------------------------------------------------------------------------
// 5x5 SAME conv as implicit GEMM with V_WMMA_F32_16X16X32_F16.
// Each wave: 16 output pixels x 32 output channels (2 WMMA n-tiles), so every
// activation fragment feeds 2 WMMAs -> 1 global b128 load per WMMA issued.
// 32-channel weight slab (100 KB) staged in LDS by the whole workgroup.
// ---------------------------------------------------------------------------
template <int RELU, int FINAL>
__global__ __launch_bounds__(256) void conv5x5_wmma(
    const half_t* __restrict__ in, const half_t* __restrict__ wts,
    const float* __restrict__ bias, half_t* __restrict__ out,
    float* __restrict__ out4, int NSB)
{
  __shared__ half_t wlds[32 * KTOT];   // 102400 bytes
  const int tid = threadIdx.x;
  const int sb  = blockIdx.x % NSB;               // 32-channel superblock
  const int yp  = (blockIdx.x / NSB) % 32;
  const int img = blockIdx.x / (NSB * 32);

  // cooperative stage of 32*KTOT halfs (12800 x b64) into LDS
  {
    const unsigned long long* src =
        (const unsigned long long*)(wts + (size_t)sb * 32 * KTOT);
    unsigned long long* dst = (unsigned long long*)wlds;
#pragma unroll
    for (int u = 0; u < 50; ++u) dst[tid + 256 * u] = src[tid + 256 * u];
  }
  __syncthreads();

  const int lane = tid & 31;
  const int wave = tid >> 5;
  const int y   = yp * 2 + (wave >> 2);   // output row 0..63
  const int x0  = (wave & 3) * 16;        // output col base
  const int l15 = lane & 15;              // A row / D column
  const int hf  = lane >> 4;              // K half selector

  const int nbase = sb * 32;
  v8f acc0, acc1;
  {
    const float b0v = bias[nbase + l15];
    const float b1v = bias[nbase + 16 + l15];
#pragma unroll
    for (int r = 0; r < 8; ++r) { acc0[r] = b0v; acc1[r] = b1v; }
  }

  const size_t imgBase = (size_t)img * PH * PW;

#pragma unroll
  for (int t = 0; t < 25; ++t) {
    const int dy = t / 5;                 // padded-coords tap offsets
    const int dx = t % 5;
    const half_t* rowp =
        in + ((imgBase + (size_t)(y + dy) * PW) + (size_t)(x0 + dx + l15)) * CINC + hf * 16;
    v16h a0, a1;
    ((float4*)&a0)[0] = *(const float4*)(rowp);        // K = hf*16 .. +7
    ((float4*)&a0)[1] = *(const float4*)(rowp + 8);    // K = hf*16+8 .. +7
    ((float4*)&a1)[0] = *(const float4*)(rowp + 32);   // second K=32 slab
    ((float4*)&a1)[1] = *(const float4*)(rowp + 40);

    const half_t* bp0 = wlds + (size_t)l15 * KTOT + t * 64 + hf * 16;          // n-tile 0
    const half_t* bp1 = bp0 + (size_t)16 * KTOT;                               // n-tile 1
    v16h b00 = *(const v16h*)(bp0);
    v16h b01 = *(const v16h*)(bp0 + 32);
    v16h b10 = *(const v16h*)(bp1);
    v16h b11 = *(const v16h*)(bp1 + 32);

    acc0 = __builtin_amdgcn_wmma_f32_16x16x32_f16(false, a0, false, b00,
                                                  (short)0, acc0, false, false);
    acc1 = __builtin_amdgcn_wmma_f32_16x16x32_f16(false, a0, false, b10,
                                                  (short)0, acc1, false, false);
    acc0 = __builtin_amdgcn_wmma_f32_16x16x32_f16(false, a1, false, b01,
                                                  (short)0, acc0, false, false);
    acc1 = __builtin_amdgcn_wmma_f32_16x16x32_f16(false, a1, false, b11,
                                                  (short)0, acc1, false, false);
  }

  if (FINAL) {
    if (l15 < 4) {
#pragma unroll
      for (int r = 0; r < 8; ++r) {
        const int mm = r + 8 * hf;
        out4[(((size_t)img * HH + y) * WWID + (x0 + mm)) * 4 + l15] = acc0[r];
      }
    }
  } else {
    half_t* op = out + ((imgBase + (size_t)(y + PADB) * PW)) * CINC;
#pragma unroll
    for (int r = 0; r < 8; ++r) {
      const int mm = r + 8 * hf;
      float v0 = acc0[r];
      float v1 = acc1[r];
      if (RELU) { v0 = v0 > 0.f ? v0 : 0.f; v1 = v1 > 0.f ? v1 : 0.f; }
      half_t* pp = op + (size_t)(x0 + mm + PADB) * CINC + nbase + l15;
      pp[0]  = (half_t)v0;
      pp[16] = (half_t)v1;
    }
  }
}

// ---------------------------------------------------------------------------
// Small helper kernels
// ---------------------------------------------------------------------------
__global__ void zero_kernel(uint4* p, size_t n) {
  size_t i = (size_t)blockIdx.x * blockDim.x + threadIdx.x;
  uint4 z = make_uint4(0u, 0u, 0u, 0u);
  for (; i < n; i += (size_t)gridDim.x * blockDim.x) p[i] = z;
}

// OIHW f32 -> [o][tap][i] f16 (o padded with zeros up to total/KTOT channels)
__global__ void prep_w_kernel(const float* __restrict__ src, half_t* __restrict__ dst,
                              int oc_valid, int total) {
  int idx = blockIdx.x * blockDim.x + threadIdx.x;
  if (idx >= total) return;
  int o = idx / KTOT;
  int rem = idx % KTOT;
  int tap = rem >> 6;
  int i = rem & 63;
  float v = 0.f;
  if (o < oc_valid) v = src[(((o * 64 + i) * 5) + tap / 5) * 5 + tap % 5];
  dst[idx] = (half_t)v;
}

__global__ void prep_b4_kernel(const float* __restrict__ src, float* __restrict__ dst) {
  int t = threadIdx.x;
  dst[t] = (t < 4) ? src[t] : 0.f;
}

__global__ void posemb_kernel(const float* __restrict__ wp, const float* __restrict__ bp,
                              float* __restrict__ pe) {
  int idx = blockIdx.x * blockDim.x + threadIdx.x;
  if (idx >= HH * WWID * DDIM) return;
  int d = idx & 63;
  int x = (idx >> 6) & 63;
  int y = idx >> 12;
  float ys = -1.f + 2.f * y / (HH - 1);
  float xs = -1.f + 2.f * x / (WWID - 1);
  pe[idx] = ys * wp[d] + xs * wp[64 + d] + bp[d];
}

// x0 = slots + pos_emb, written as padded NHWC f16
__global__ void build_x_kernel(const float* __restrict__ cur, const float* __restrict__ pe,
                               half_t* __restrict__ act) {
  int idx = blockIdx.x * blockDim.x + threadIdx.x;
  int d = idx & 63;
  int x = (idx >> 6) & 63;
  int y = (idx >> 12) & 63;
  int bk = idx >> 18;
  if (bk >= NIMG) return;
  float v = cur[bk * DDIM + d] + pe[((y << 6) + x) * 64 + d];
  act[(((size_t)bk * PH + (y + PADB)) * PW + (x + PADB)) * CINC + d] = (half_t)v;
}

// softmax over K slots on channel 3, weighted RGB sum -> frame f
__global__ void combine_kernel(const float* __restrict__ out4, float* __restrict__ out,
                               int f) {
  int idx = blockIdx.x * blockDim.x + threadIdx.x;
  if (idx >= BBATCH * HH * WWID) return;
  int b = idx >> 12;
  int p = idx & 4095;
  float ml[KSLOT];
  float mx = -1e30f;
  for (int k = 0; k < KSLOT; ++k) {
    ml[k] = out4[(((size_t)(b * KSLOT + k) << 12) + p) * 4 + 3];
    mx = fmaxf(mx, ml[k]);
  }
  float se = 0.f;
  for (int k = 0; k < KSLOT; ++k) { ml[k] = __expf(ml[k] - mx); se += ml[k]; }
  float inv = 1.f / se;
  for (int c = 0; c < 3; ++c) {
    float a = 0.f;
    for (int k = 0; k < KSLOT; ++k)
      a += out4[(((size_t)(b * KSLOT + k) << 12) + p) * 4 + c] * ml[k];
    out[(((size_t)b * NFRAME + f) * 3 + c) * 4096 + p] = a * inv;
  }
}

// gating + interaction MLPs; block = (b, j); accumulate over i in registers
__global__ __launch_bounds__(64) void interact_kernel(
    const float* __restrict__ cur, float* __restrict__ di,
    const float* __restrict__ wg1, const float* __restrict__ bg1,
    const float* __restrict__ wg2, const float* __restrict__ bg2,
    const float* __restrict__ wge, const float* __restrict__ bge,
    const float* __restrict__ wi1, const float* __restrict__ bi1,
    const float* __restrict__ wi2, const float* __restrict__ bi2) {
  __shared__ float pairb[128];
  __shared__ float hbuf[64];
  __shared__ float red[64];
  const int t = threadIdx.x;
  const int b = blockIdx.x / KSLOT;
  const int j = blockIdx.x % KSLOT;
  const float sj = cur[(b * KSLOT + j) * DDIM + t];
  float acc = 0.f;
  for (int i = 0; i < KSLOT; ++i) {
    __syncthreads();
    pairb[t] = cur[(b * KSLOT + i) * DDIM + t];
    pairb[64 + t] = sj;
    __syncthreads();
    float h1 = bg1[t];
    for (int k = 0; k < 128; ++k) h1 += pairb[k] * wg1[k * 64 + t];
    h1 = fmaxf(h1, 0.f);
    hbuf[t] = h1;
    __syncthreads();
    float h2 = bg2[t];
    for (int k = 0; k < 64; ++k) h2 += hbuf[k] * wg2[k * 64 + t];
    h2 = fmaxf(h2, 0.f);
    red[t] = h2 * wge[t];
    __syncthreads();
    for (int s = 32; s > 0; s >>= 1) {
      if (t < s) red[t] += red[t + s];
      __syncthreads();
    }
    float ep = 1.f / (1.f + __expf(-(red[0] + bge[0])));
    __syncthreads();
    float m1 = bi1[t];
    for (int k = 0; k < 128; ++k) m1 += pairb[k] * wi1[k * 64 + t];
    m1 = fmaxf(m1, 0.f);
    hbuf[t] = m1;
    __syncthreads();
    float m2 = bi2[t];
    for (int k = 0; k < 64; ++k) m2 += hbuf[k] * wi2[k * 64 + t];
    acc += m2 * ep;
  }
  di[(b * KSLOT + j) * DDIM + t] = acc;
}

// LN(cur) -> self MLP; u=[ds,di]; LN(u) -> update MLP; in-place cur update
__global__ __launch_bounds__(64) void update_kernel(
    float* __restrict__ cur, const float* __restrict__ di,
    const float* __restrict__ lnsg, const float* __restrict__ lnsb,
    const float* __restrict__ ws1, const float* __restrict__ bs1,
    const float* __restrict__ ws2, const float* __restrict__ bs2,
    const float* __restrict__ lnug, const float* __restrict__ lnub,
    const float* __restrict__ wu1, const float* __restrict__ bu1,
    const float* __restrict__ wu2, const float* __restrict__ bu2) {
  __shared__ float xb[128];
  __shared__ float hb[64];
  __shared__ float red[64];
  const int t = threadIdx.x;
  const int bi = blockIdx.x;
  float x = cur[bi * DDIM + t];
  red[t] = x;
  __syncthreads();
  for (int s = 32; s > 0; s >>= 1) { if (t < s) red[t] += red[t + s]; __syncthreads(); }
  float mean = red[0] * (1.f / 64.f);
  __syncthreads();
  red[t] = (x - mean) * (x - mean);
  __syncthreads();
  for (int s = 32; s > 0; s >>= 1) { if (t < s) red[t] += red[t + s]; __syncthreads(); }
  float var = red[0] * (1.f / 64.f);
  float xln = (x - mean) * rsqrtf(var + 1e-5f) * lnsg[t] + lnsb[t];
  __syncthreads();
  xb[t] = xln;
  __syncthreads();
  float h = bs1[t];
  for (int k = 0; k < 64; ++k) h += xb[k] * ws1[k * 64 + t];
  h = fmaxf(h, 0.f);
  hb[t] = h;
  __syncthreads();
  float dsv = bs2[t];
  for (int k = 0; k < 64; ++k) dsv += hb[k] * ws2[k * 64 + t];
  __syncthreads();
  xb[t] = dsv;
  xb[64 + t] = di[bi * DDIM + t];
  __syncthreads();
  red[t] = xb[t] + xb[64 + t];
  __syncthreads();
  for (int s = 32; s > 0; s >>= 1) { if (t < s) red[t] += red[t + s]; __syncthreads(); }
  float mu = red[0] * (1.f / 128.f);
  __syncthreads();
  float d0 = xb[t] - mu, d1 = xb[64 + t] - mu;
  red[t] = d0 * d0 + d1 * d1;
  __syncthreads();
  for (int s = 32; s > 0; s >>= 1) { if (t < s) red[t] += red[t + s]; __syncthreads(); }
  float rs = rsqrtf(red[0] * (1.f / 128.f) + 1e-5f);
  __syncthreads();
  xb[t]      = d0 * rs * lnug[t] + lnub[t];
  xb[64 + t] = d1 * rs * lnug[64 + t] + lnub[64 + t];
  __syncthreads();
  float h2 = bu1[t];
  for (int k = 0; k < 128; ++k) h2 += xb[k] * wu1[k * 64 + t];
  h2 = fmaxf(h2, 0.f);
  hb[t] = h2;
  __syncthreads();
  float nv = bu2[t];
  for (int k = 0; k < 64; ++k) nv += hb[k] * wu2[k * 64 + t];
  cur[bi * DDIM + t] = nv;
}

// ---------------------------------------------------------------------------
extern "C" void kernel_launch(void* const* d_in, const int* in_sizes, int n_in,
                              void* d_out, int out_size, void* d_ws, size_t ws_size,
                              hipStream_t stream) {
  (void)in_sizes; (void)n_in; (void)out_size; (void)ws_size;
  const float* slots = (const float*)d_in[0];
  const float* lnsg = (const float*)d_in[1];
  const float* lnsb = (const float*)d_in[2];
  const float* ws1v = (const float*)d_in[3];
  const float* bs1v = (const float*)d_in[4];
  const float* ws2v = (const float*)d_in[5];
  const float* bs2v = (const float*)d_in[6];
  const float* wi1v = (const float*)d_in[7];
  const float* bi1v = (const float*)d_in[8];
  const float* wi2v = (const float*)d_in[9];
  const float* bi2v = (const float*)d_in[10];
  const float* lnug = (const float*)d_in[11];
  const float* lnub = (const float*)d_in[12];
  const float* wu1v = (const float*)d_in[13];
  const float* bu1v = (const float*)d_in[14];
  const float* wu2v = (const float*)d_in[15];
  const float* bu2v = (const float*)d_in[16];
  const float* wg1v = (const float*)d_in[17];
  const float* bg1v = (const float*)d_in[18];
  const float* wg2v = (const float*)d_in[19];
  const float* bg2v = (const float*)d_in[20];
  const float* wgev = (const float*)d_in[21];
  const float* bgev = (const float*)d_in[22];
  const float* wpv  = (const float*)d_in[23];
  const float* bpv  = (const float*)d_in[24];
  const float* wc1  = (const float*)d_in[25];
  const float* bc1  = (const float*)d_in[26];
  const float* wc2  = (const float*)d_in[27];
  const float* bc2  = (const float*)d_in[28];
  const float* wc3  = (const float*)d_in[29];
  const float* bc3  = (const float*)d_in[30];
  const float* wc4  = (const float*)d_in[31];
  const float* bc4  = (const float*)d_in[32];
  float* out = (float*)d_out;

  const size_t ACT_BYTES = (size_t)NIMG * PH * PW * CINC * sizeof(half_t); // 56.8 MB
  char* ws = (char*)d_ws;
  size_t off = 0;
  auto alloc = [&](size_t bytes) -> void* {
    void* p = ws + off;
    off += (bytes + 255) & ~(size_t)255;
    return p;
  };
  half_t* A0 = (half_t*)alloc(ACT_BYTES);
  half_t* A1 = (half_t*)alloc(ACT_BYTES);
  half_t* W1 = (half_t*)alloc((size_t)64 * KTOT * sizeof(half_t));
  half_t* W2 = (half_t*)alloc((size_t)64 * KTOT * sizeof(half_t));
  half_t* W3 = (half_t*)alloc((size_t)64 * KTOT * sizeof(half_t));
  half_t* W4 = (half_t*)alloc((size_t)32 * KTOT * sizeof(half_t));  // padded to 32 oc
  float* bias4 = (float*)alloc(32 * sizeof(float));
  float* pe    = (float*)alloc((size_t)HH * WWID * DDIM * sizeof(float));
  float* cur   = (float*)alloc((size_t)BBATCH * KSLOT * DDIM * sizeof(float));
  float* di    = (float*)alloc((size_t)BBATCH * KSLOT * DDIM * sizeof(float));
  float* out4  = (float*)alloc((size_t)NIMG * HH * WWID * 4 * sizeof(float));

  // zero both activation buffers once (A0,A1 are contiguous); borders stay 0
  zero_kernel<<<4096, 256, 0, stream>>>((uint4*)A0, (2 * ACT_BYTES) / 16);

  prep_w_kernel<<<(64 * KTOT + 255) / 256, 256, 0, stream>>>(wc1, W1, 64, 64 * KTOT);
  prep_w_kernel<<<(64 * KTOT + 255) / 256, 256, 0, stream>>>(wc2, W2, 64, 64 * KTOT);
  prep_w_kernel<<<(64 * KTOT + 255) / 256, 256, 0, stream>>>(wc3, W3, 64, 64 * KTOT);
  prep_w_kernel<<<(32 * KTOT + 255) / 256, 256, 0, stream>>>(wc4, W4, 4, 32 * KTOT);
  prep_b4_kernel<<<1, 32, 0, stream>>>(bc4, bias4);
  posemb_kernel<<<(HH * WWID * DDIM) / 256, 256, 0, stream>>>(wpv, bpv, pe);
  hipMemcpyAsync(cur, slots, (size_t)BBATCH * KSLOT * DDIM * sizeof(float),
                 hipMemcpyDeviceToDevice, stream);

  for (int f = 0; f < NFRAME; ++f) {
    build_x_kernel<<<(NIMG * HH * WWID * DDIM) / 256, 256, 0, stream>>>(cur, pe, A0);
    conv5x5_wmma<1, 0><<<NIMG * 32 * 2, 256, 0, stream>>>(A0, W1, bc1, A1, nullptr, 2);
    conv5x5_wmma<1, 0><<<NIMG * 32 * 2, 256, 0, stream>>>(A1, W2, bc2, A0, nullptr, 2);
    conv5x5_wmma<1, 0><<<NIMG * 32 * 2, 256, 0, stream>>>(A0, W3, bc3, A1, nullptr, 2);
    conv5x5_wmma<0, 1><<<NIMG * 32 * 1, 256, 0, stream>>>(A1, W4, bias4, nullptr, out4, 1);
    combine_kernel<<<(BBATCH * HH * WWID + 255) / 256, 256, 0, stream>>>(out4, out, f);
    if (f < NFRAME - 1) {
      interact_kernel<<<BBATCH * KSLOT, 64, 0, stream>>>(
          cur, di, wg1v, bg1v, wg2v, bg2v, wgev, bgev, wi1v, bi1v, wi2v, bi2v);
      update_kernel<<<BBATCH * KSLOT, 64, 0, stream>>>(
          cur, di, lnsg, lnsb, ws1v, bs1v, ws2v, bs2v, lnug, lnub, wu1v, bu1v, wu2v, bu2v);
    }
  }
}
